// xLSTMCell_24601572671658
// MI455X (gfx1250) — matrix operand order repaired and checked by reading, not compiled, not run on hardware
//
#include <hip/hip_runtime.h>

// ---------------------------------------------------------------------------
// xLSTM cell on MI455X (gfx1250, wave32, WMMA).
//
// One GEMM:  A=[x|h] (8192x2048)  x  Vg=[W|U] (4x1024x2048)^T, 4 gates fused.
// f32 inputs split into bf16 hi/lo (v = hi + lo); each K=32 step does
// 3x v_wmma_f32_16x16x32_bf16 (hi*hi, hi*lo, lo*hi) per tile, f32 accumulate.
//
// Workspace layout interleaves hi/lo per row so lo sits at +4096 B (imm24):
//   xh  [8192][2][2048] u16   (part 0 = hi, part 1 = lo)   64 MB
//   wgt [4096][2][2048] u16                                32 MB
// Hot loop: 1 A-pointer + 4 B-pointers per K-step; all 40 b128 loads use
// constant immediates off those 5 vaddrs; 48 WMMAs per K-step (32 rows/wave).
// ---------------------------------------------------------------------------

typedef __attribute__((ext_vector_type(16))) __bf16 v16bf;
typedef __attribute__((ext_vector_type(8)))  float  v8f;

#define BATCH 8192
#define HDIM  1024
#define KDIM  2048            // I + H
#define RSTRIDE 4096          // elements per row in interleaved layout (2*KDIM)
#define BH    (BATCH * HDIM)  // 8388608

// ---- f32 -> bf16 hi/lo split (hi = truncate, lo = RNE of residual) --------
__device__ __forceinline__ void split_bf16(float v, unsigned short& hi, unsigned short& lo) {
    unsigned int u  = __float_as_uint(v);
    unsigned int hb = u & 0xFFFF0000u;
    float hf = __uint_as_float(hb);
    float lf = v - hf;                       // exact residual
    hi = (unsigned short)(hb >> 16);
    unsigned int ul = __float_as_uint(lf);
    unsigned int r  = ul + 0x7FFFu + ((ul >> 16) & 1u);   // round-to-nearest-even
    lo = (unsigned short)(r >> 16);
}

// ---- pack [x | h] -> xh (hi plane + lo plane per row) ----------------------
__global__ __launch_bounds__(256) void pack_xh(const float* __restrict__ x,
                                               const float* __restrict__ h,
                                               unsigned short* __restrict__ xh) {
    unsigned int idx = blockIdx.x * 256u + threadIdx.x;    // 0 .. 8192*2048-1
    unsigned int b = idx >> 11;
    unsigned int k = idx & 2047u;
    float v = (k < 1024u) ? x[b * 1024u + k] : h[b * 1024u + (k - 1024u)];
    unsigned short hi, lo;
    split_bf16(v, hi, lo);
    xh[b * RSTRIDE + k]        = hi;
    xh[b * RSTRIDE + KDIM + k] = lo;
}

// ---- pack [W | U] per gate-row -> wgt (hi plane + lo plane per row) --------
__global__ __launch_bounds__(256) void pack_w(const float* __restrict__ W,
                                              const float* __restrict__ U,
                                              unsigned short* __restrict__ wg) {
    unsigned int idx = blockIdx.x * 256u + threadIdx.x;    // 0 .. 4096*2048-1
    unsigned int row = idx >> 11;                          // g*1024 + n
    unsigned int k   = idx & 2047u;
    float v = (k < 1024u) ? W[row * 1024u + k] : U[row * 1024u + (k - 1024u)];
    unsigned short hi, lo;
    split_bf16(v, hi, lo);
    wg[row * RSTRIDE + k]        = hi;
    wg[row * RSTRIDE + KDIM + k] = lo;
}

// ---- fragment loaders (per ISA 16-bit A 16x32 and B 32x16 layouts) ---------
// A: lane L<16 holds row M=L, K = {c..c+7, c+16..c+23}; lanes 16..31 use c+8.
//    p already includes the 8*(lane>>4) shift: two 16B chunks, +32B apart.
__device__ __forceinline__ v16bf load_frag_a(const unsigned short* p) {
    union { uint4 q[2]; v16bf v; } u;
    u.q[0] = *reinterpret_cast<const uint4*>(p);        // elems 0..7
    u.q[1] = *reinterpret_cast<const uint4*>(p + 16);   // elems 8..15 (K +16)
    return u.v;
}
// B: lane holds column N=lane&15, K = {c..c+7, c+16..c+23}, c = k0 + 8*(lane>>4).
__device__ __forceinline__ v16bf load_frag_b(const unsigned short* p) {
    union { uint4 q[2]; v16bf v; } u;
    u.q[0] = *reinterpret_cast<const uint4*>(p);
    u.q[1] = *reinterpret_cast<const uint4*>(p + 16);
    return u.v;
}

__device__ __forceinline__ v8f wmma_bf16(v16bf a, v16bf b, v8f c) {
    return __builtin_amdgcn_wmma_f32_16x16x32_bf16(false, a, false, b, (short)0, c, false, false);
}

__device__ __forceinline__ float sigmoid_f(float x) { return 1.0f / (1.0f + __expf(-x)); }
__device__ __forceinline__ float tanh_f(float x) {
    x = fminf(fmaxf(x, -15.0f), 15.0f);
    float e = __expf(2.0f * x);
    return (e - 1.0f) / (e + 1.0f);
}

// ---- fused GEMM + LSTM cell ------------------------------------------------
// grid: (HDIM/32, BATCH/256); block: 256 threads = 8 waves.
// wave w: rows [by*256 + w*32, +32) (2 m-tiles), cols [bx*32, +32) (2 n-tiles),
// all 4 gates -> 16 accumulators, 48 WMMAs per K=32 step.
__global__ __launch_bounds__(256) void gemm_lstm(const unsigned short* __restrict__ xh,
                                                 const unsigned short* __restrict__ wg,
                                                 const float* __restrict__ c_in,
                                                 const float* __restrict__ bW,
                                                 const float* __restrict__ bU,
                                                 float* __restrict__ out) {
    const int lane  = threadIdx.x & 31;
    const int wid   = threadIdx.x >> 5;
    const int laneh = lane >> 4;          // 0 for lanes 0-15, 1 for 16-31
    const int lanel = lane & 15;

    const int m0 = blockIdx.y * 256 + wid * 32;   // wave row base (2 m-tiles)
    const int n0 = blockIdx.x * 32;               // block col base (2 n-tiles)

    v8f acc[4][2][2] = {};                        // [gate][ntile][mtile]

    // Loop-invariant per-lane element offsets.
    const unsigned int a_voff = (unsigned int)(m0 + lanel) * RSTRIDE + 8u * laneh;
    unsigned int b_voff[4];
#pragma unroll
    for (int g = 0; g < 4; ++g)
        b_voff[g] = (unsigned int)(g * 1024 + n0 + lanel) * RSTRIDE + 8u * laneh;

    // Constant element offsets (all fit the signed-24-bit byte immediate):
    //   lo plane: +KDIM (=4096 B); next 16-row sub-tile: +16*RSTRIDE (=131072 B)
    for (int k0 = 0; k0 < KDIM; k0 += 32) {
        const unsigned short* xh_k = xh + k0;     // scalar base, s_add per iter
        const unsigned short* wg_k = wg + k0;

        const unsigned short* pa = xh_k + a_voff;
        v16bf ah0 = load_frag_a(pa);                          // m-tile 0, hi
        v16bf al0 = load_frag_a(pa + KDIM);                   // m-tile 0, lo
        v16bf ah1 = load_frag_a(pa + 16 * RSTRIDE);           // m-tile 1, hi
        v16bf al1 = load_frag_a(pa + 16 * RSTRIDE + KDIM);    // m-tile 1, lo
#pragma unroll
        for (int g = 0; g < 4; ++g) {
            const unsigned short* pb = wg_k + b_voff[g];
            v16bf bh0 = load_frag_b(pb);                        // n-tile 0, hi
            v16bf bl0 = load_frag_b(pb + KDIM);                 // n-tile 0, lo
            v16bf bh1 = load_frag_b(pb + 16 * RSTRIDE);         // n-tile 1, hi
            v16bf bl1 = load_frag_b(pb + 16 * RSTRIDE + KDIM);  // n-tile 1, lo
            // n-tile 0: each B fragment feeds both m-tiles
            acc[g][0][0] = wmma_bf16(ah0, bh0, acc[g][0][0]);
            acc[g][0][1] = wmma_bf16(ah1, bh0, acc[g][0][1]);
            acc[g][0][0] = wmma_bf16(ah0, bl0, acc[g][0][0]);
            acc[g][0][1] = wmma_bf16(ah1, bl0, acc[g][0][1]);
            acc[g][0][0] = wmma_bf16(al0, bh0, acc[g][0][0]);
            acc[g][0][1] = wmma_bf16(al1, bh0, acc[g][0][1]);
            // n-tile 1
            acc[g][1][0] = wmma_bf16(ah0, bh1, acc[g][1][0]);
            acc[g][1][1] = wmma_bf16(ah1, bh1, acc[g][1][1]);
            acc[g][1][0] = wmma_bf16(ah0, bl1, acc[g][1][0]);
            acc[g][1][1] = wmma_bf16(ah1, bl1, acc[g][1][1]);
            acc[g][1][0] = wmma_bf16(al0, bh1, acc[g][1][0]);
            acc[g][1][1] = wmma_bf16(al1, bh1, acc[g][1][1]);
        }
    }

    // ---- fused LSTM epilogue ----
    // C/D layout: lane 0-15 -> N=lane, M=r; lane 16-31 -> N=lane-16, M=8+r.
#pragma unroll
    for (int nt = 0; nt < 2; ++nt) {
        const int n = n0 + nt * 16 + lanel;
        float bia[4];
#pragma unroll
        for (int g = 0; g < 4; ++g) bia[g] = bW[g * 1024 + n] + bU[g * 1024 + n];
#pragma unroll
        for (int mt = 0; mt < 2; ++mt) {
#pragma unroll
            for (int r = 0; r < 8; ++r) {
                const int m = m0 + mt * 16 + r + 8 * laneh;
                const float gi = acc[0][nt][mt][r] + bia[0];
                const float gf = acc[1][nt][mt][r] + bia[1];
                const float go = acc[2][nt][mt][r] + bia[2];
                const float gc = acc[3][nt][mt][r] + bia[3];
                const float i_ = sigmoid_f(gi);
                const float f_ = sigmoid_f(gf);
                const float o_ = sigmoid_f(go);
                const float ch = tanh_f(gc);
                const unsigned int idx = (unsigned int)m * HDIM + n;
                const float cn = f_ * __builtin_nontemporal_load(&c_in[idx]) + i_ * ch;
                const float hn = o_ * tanh_f(cn);
                __builtin_nontemporal_store(hn, &out[idx]);           // h_new
                __builtin_nontemporal_store(hn, &out[idx + BH]);      // h_new (tuple)
                __builtin_nontemporal_store(cn, &out[idx + 2 * BH]);  // c_new
            }
        }
    }
}

// ---------------------------------------------------------------------------
extern "C" void kernel_launch(void* const* d_in, const int* in_sizes, int n_in,
                              void* d_out, int out_size, void* d_ws, size_t ws_size,
                              hipStream_t stream) {
    const float* x  = (const float*)d_in[0];
    const float* h  = (const float*)d_in[1];
    const float* c  = (const float*)d_in[2];
    const float* W  = (const float*)d_in[3];
    const float* bW = (const float*)d_in[4];
    const float* U  = (const float*)d_in[5];
    const float* bU = (const float*)d_in[6];
    float* out = (float*)d_out;

    // workspace carve-up (needs 96 MB)
    unsigned short* xh = (unsigned short*)d_ws;                     // 64 MB
    unsigned short* wg = xh + (size_t)BATCH * RSTRIDE;              // 32 MB

    pack_xh<<<(BATCH * KDIM) / 256, 256, 0, stream>>>(x, h, xh);
    pack_w <<<(4 * HDIM * KDIM) / 256, 256, 0, stream>>>(W, U, wg);

    dim3 grid(HDIM / 32, BATCH / 256);
    gemm_lstm<<<grid, 256, 0, stream>>>(xh, wg, c, bW, bU, out);
}
